// Decoder_Block_72705206386870
// MI455X (gfx1250) — matrix operand order repaired and checked by reading, not compiled, not run on hardware
//
#include <hip/hip_runtime.h>
#include <math.h>

typedef __attribute__((ext_vector_type(16))) _Float16 v16h;
typedef __attribute__((ext_vector_type(8)))  _Float16 v8h;
typedef __attribute__((ext_vector_type(8)))  float    v8f;

#define LN_EPS 1e-5f

enum { EP_NONE = 0, EP_BNRELU = 1, EP_BIAS = 2, EP_BIAS_GELU = 3, EP_BIAS_ADD = 4, EP_ADD = 5 };

__device__ __forceinline__ float siluf(float x) { return x / (1.f + expf(-x)); }
__device__ __forceinline__ float geluf(float x) {
  return 0.5f * x * (1.f + tanhf(0.7978845608f * (x + 0.044715f * x * x * x)));
}
__device__ __forceinline__ float softplusf(float x) {
  return (x > 20.f) ? x : log1pf(expf(x));
}

// ---------------------------------------------------------------------------
// Generic WMMA GEMM:  C[z] = epilogue(A[z'] (MxK, row-major) * B[z] (KxN, row-major))
// 64x64 block tile, 4 waves of 32x32 each, K-step 32, f32->f16 staged in LDS.
// ---------------------------------------------------------------------------
#define GBM 64
#define GBN 64
#define GBK 32
#define GLS 40  // LDS row stride in halves (80B: keeps 16B alignment, skews banks)

__global__ __launch_bounds__(128) void dk_gemm(
    const float* __restrict__ A, int lda, long long sAz, int azm,
    const float* __restrict__ Bm, int ldb, long long sBz,
    float* __restrict__ C, int ldc, long long sCz,
    int M, int N, int K, int ep,
    const float* __restrict__ p0, const float* __restrict__ p1,
    const float* __restrict__ addp, int ldadd, long long sAddz) {
  __shared__ _Float16 sA[GBM * GLS];
  __shared__ _Float16 sB[GBN * GLS];

  int z = blockIdx.z;
  const float* Ab = A + (long long)(azm ? (z % azm) : z) * sAz;
  const float* Bb = Bm + (long long)z * sBz;
  float* Cb = C + (long long)z * sCz;

  int m0 = blockIdx.y * GBM;
  int n0 = blockIdx.x * GBN;
  int tid = threadIdx.x;
  int lane = tid & 31;
  int wv = tid >> 5;
  int wr = wv >> 1;  // wave row (0..1)
  int wc = wv & 1;   // wave col (0..1)

  v8f acc[2][2];
#pragma unroll
  for (int i = 0; i < 2; ++i)
#pragma unroll
    for (int j = 0; j < 2; ++j)
#pragma unroll
      for (int t = 0; t < 8; ++t) acc[i][j][t] = 0.f;

  for (int k0 = 0; k0 < K; k0 += GBK) {
    // Stage A tile (64x32) as f16, zero padded at tails.
#pragma unroll
    for (int t = 0; t < 16; ++t) {
      int e = tid + t * 128;
      int r = e >> 5, c = e & 31;
      int gm = m0 + r, gk = k0 + c;
      float v = (gm < M && gk < K) ? Ab[(long long)gm * lda + gk] : 0.f;
      sA[r * GLS + c] = (_Float16)v;
    }
    // Stage B tile (32x64) transposed into [n][k] so fragments are contiguous.
#pragma unroll
    for (int t = 0; t < 16; ++t) {
      int e = tid + t * 128;
      int kk = e >> 6, n = e & 63;
      int gk = k0 + kk, gn = n0 + n;
      float v = (gk < K && gn < N) ? Bb[(long long)gk * ldb + gn] : 0.f;
      sB[n * GLS + kk] = (_Float16)v;
    }
    __syncthreads();

    // A fragments: lanes 0-15 hold K {0..7,16..23}, lanes 16-31 hold {8..15,24..31}.
    int kba = (lane < 16) ? 0 : 8;
    int mrb = wr * 32 + (lane & 15);
    v16h afrag[2];
#pragma unroll
    for (int i = 0; i < 2; ++i) {
      const _Float16* pa = &sA[(mrb + i * 16) * GLS + kba];
      v8h a0 = *(const v8h*)pa;
      v8h a1 = *(const v8h*)(pa + 16);
#pragma unroll
      for (int t = 0; t < 8; ++t) { afrag[i][t] = a0[t]; afrag[i][8 + t] = a1[t]; }
    }
    // B fragments: lanes 0-15 hold K {0..7,16..23}, lanes 16-31 hold {8..15,24..31}.
    int kbb = (lane < 16) ? 0 : 8;
    int ncb = wc * 32 + (lane & 15);
    v16h bfrag[2];
#pragma unroll
    for (int j = 0; j < 2; ++j) {
      const _Float16* pb = &sB[(ncb + j * 16) * GLS + kbb];
      v8h b0 = *(const v8h*)pb;
      v8h b1 = *(const v8h*)(pb + 16);
#pragma unroll
      for (int t = 0; t < 8; ++t) { bfrag[j][t] = b0[t]; bfrag[j][8 + t] = b1[t]; }
    }

#pragma unroll
    for (int i = 0; i < 2; ++i)
#pragma unroll
      for (int j = 0; j < 2; ++j)
        acc[i][j] = __builtin_amdgcn_wmma_f32_16x16x32_f16(
            false, afrag[i], false, bfrag[j], (short)0, acc[i][j], false, false);
    __syncthreads();
  }

  // Epilogue + store (C/D layout: VGPR r -> row r (+8 for lanes 16-31), col = lane&15).
  int rowadd = (lane < 16) ? 0 : 8;
  int coll = lane & 15;
#pragma unroll
  for (int i = 0; i < 2; ++i)
#pragma unroll
    for (int j = 0; j < 2; ++j)
#pragma unroll
      for (int r = 0; r < 8; ++r) {
        int gm = m0 + wr * 32 + i * 16 + r + rowadd;
        int gn = n0 + wc * 32 + j * 16 + coll;
        if (gm < M && gn < N) {
          float v = acc[i][j][r];
          if (ep == EP_BNRELU) {
            v = v * p0[gm] + p1[gm];
            v = fmaxf(v, 0.f);
          } else if (ep == EP_BIAS) {
            v += p0[gn];
          } else if (ep == EP_BIAS_GELU) {
            v = geluf(v + p0[gn]);
          } else if (ep == EP_BIAS_ADD) {
            v += p0[gn] + addp[(long long)z * sAddz + (long long)gm * ldadd + gn];
          } else if (ep == EP_ADD) {
            v += addp[(long long)z * sAddz + (long long)gm * ldadd + gn];
          }
          Cb[(long long)gm * ldc + gn] = v;
        }
      }
}

// ---------------------------------------------------------------------------
// BN prep: scale = g*rsqrt(v+eps), shift = b - m*scale
// ---------------------------------------------------------------------------
__global__ void dk_bnprep(const float* g, const float* b, const float* m,
                          const float* v, float* scale, float* shift, int n) {
  int c = blockIdx.x * blockDim.x + threadIdx.x;
  if (c < n) {
    float s = g[c] * rsqrtf(v[c] + LN_EPS);
    scale[c] = s;
    shift[c] = b[c] - m[c] * s;
  }
}

// ---------------------------------------------------------------------------
// LayerNorm over last dim; one wave per token; optional fused silu gate.
// ---------------------------------------------------------------------------
__global__ __launch_bounds__(256) void dk_layernorm(
    const float* __restrict__ x, float* __restrict__ y, int T, int D,
    const float* __restrict__ g, const float* __restrict__ b,
    const float* __restrict__ gate, int gld, int goff) {
  int wid = threadIdx.x >> 5, lane = threadIdx.x & 31;
  int tok = blockIdx.x * (blockDim.x >> 5) + wid;
  if (tok >= T) return;
  const float* xr = x + (long long)tok * D;
  float s = 0.f, s2 = 0.f;
  for (int c = lane; c < D; c += 32) {
    float v = xr[c];
    s += v;
    s2 += v * v;
  }
#pragma unroll
  for (int o = 16; o; o >>= 1) {
    s += __shfl_xor(s, o, 32);
    s2 += __shfl_xor(s2, o, 32);
  }
  float mean = s / (float)D;
  float var = s2 / (float)D - mean * mean;
  float rstd = rsqrtf(var + LN_EPS);
  for (int c = lane; c < D; c += 32) {
    float v = (xr[c] - mean) * rstd * g[c] + b[c];
    if (gate) v *= siluf(gate[(long long)tok * gld + goff + c]);
    y[(long long)tok * D + c] = v;
  }
}

// ---------------------------------------------------------------------------
// Layout shufflers
// ---------------------------------------------------------------------------
__global__ void dk_nchw_to_tok(const float* __restrict__ s, float* __restrict__ d,
                               int B, int Cd, int H, int W) {
  int n = B * Cd * H * W;
  int i = blockIdx.x * blockDim.x + threadIdx.x;
  if (i >= n) return;
  int w = i % W, e = i / W;
  int h = e % H; e /= H;
  int c = e % Cd; e /= Cd;
  int b = e;
  d[(((long long)b * H + h) * W + w) * Cd + c] = s[i];
}

__global__ void dk_tok_to_nchw(const float* __restrict__ s, float* __restrict__ d,
                               int B, int Cd, int H, int W) {
  int n = B * Cd * H * W;
  int i = blockIdx.x * blockDim.x + threadIdx.x;
  if (i >= n) return;
  int w = i % W, e = i / W;
  int h = e % H; e /= H;
  int c = e % Cd; e /= Cd;
  int b = e;
  d[i] = s[(((long long)b * H + h) * W + w) * Cd + c];
}

// Split xz (tokens x 2di) -> first di channels as NCHW
__global__ void dk_split_nchw(const float* __restrict__ xz, float* __restrict__ d,
                              int B, int di, int H, int W) {
  int n = B * di * H * W;
  int i = blockIdx.x * blockDim.x + threadIdx.x;
  if (i >= n) return;
  int w = i % W, e = i / W;
  int h = e % H; e /= H;
  int c = e % di; e /= di;
  int b = e;
  d[i] = xz[(((long long)b * H + h) * W + w) * (2 * di) + c];
}

// Depthwise 3x3 (SAME) + bias + silu
__global__ void dk_dwconv_silu(const float* __restrict__ x, const float* __restrict__ cw,
                               const float* __restrict__ cb, float* __restrict__ y,
                               int B, int Cd, int H, int W) {
  int n = B * Cd * H * W;
  int i = blockIdx.x * blockDim.x + threadIdx.x;
  if (i >= n) return;
  int w = i % W, e = i / W;
  int h = e % H; e /= H;
  int c = e % Cd; e /= Cd;
  int b = e;
  const float* xp = x + ((long long)(b * Cd + c) * H) * W;
  float acc = cb[c];
#pragma unroll
  for (int ky = 0; ky < 3; ++ky) {
    int hh = h + ky - 1;
    if (hh < 0 || hh >= H) continue;
#pragma unroll
    for (int kx = 0; kx < 3; ++kx) {
      int ww = w + kx - 1;
      if (ww < 0 || ww >= W) continue;
      acc += xp[hh * W + ww] * cw[c * 9 + ky * 3 + kx];
    }
  }
  y[i] = siluf(acc);
}

// Build xs[b, k(0..3), d, L] (4 scan directions) from xconv NCHW
__global__ void dk_build_xs(const float* __restrict__ xc, float* __restrict__ xs,
                            int B, int di, int H, int W) {
  int n = B * di * H * W;
  int i = blockIdx.x * blockDim.x + threadIdx.x;
  if (i >= n) return;
  int L = H * W;
  int w = i % W, e = i / W;
  int h = e % H; e /= H;
  int c = e % di; e /= di;
  int b = e;
  float v = xc[i];
  int l0 = h * W + w;
  int l1 = w * H + h;
  long long base = ((long long)b * 4) * di * L + (long long)c * L;
  long long stride = (long long)di * L;
  xs[base + l0] = v;
  xs[base + stride + l1] = v;
  xs[base + 2 * stride + (L - 1 - l0)] = v;
  xs[base + 3 * stride + (L - 1 - l1)] = v;
}

// dts = softplus(dts + dt_b[k,d]) in place, layout [B,4,di,L]
__global__ void dk_softplus_dt(float* __restrict__ dts, const float* __restrict__ dtb,
                               int B, int di, int L) {
  long long n = (long long)B * 4 * di * L;
  long long i = (long long)blockIdx.x * blockDim.x + threadIdx.x;
  if (i >= n) return;
  long long e = i / L;
  int d = (int)(e % di);
  int k = (int)((e / di) % 4);
  dts[i] = softplusf(dts[i] + dtb[k * di + d]);
}

// Selective scan: one thread per (b,k,d) channel, sequential over L, N=8 state.
__global__ void dk_scan(const float* __restrict__ xs, const float* __restrict__ dts,
                        const float* __restrict__ xdbl, const float* __restrict__ Alog,
                        const float* __restrict__ Ds, float* __restrict__ ys,
                        int B, int di, int L, int R, int Cdim) {
  int i = blockIdx.x * blockDim.x + threadIdx.x;
  int total = B * 4 * di;
  if (i >= total) return;
  int d = i % di;
  int k = (i / di) % 4;
  int b = i / (4 * di);
  long long ch = (long long)i * L;
  const float* xp = xs + ch;
  const float* dp = dts + ch;
  float* yp = ys + ch;
  const float* bk = xdbl + ((long long)(b * 4 + k) * Cdim) * L;
  const float* Bs = bk + (long long)R * L;
  const float* Cs = bk + (long long)(R + 8) * L;
  float A[8], h[8];
#pragma unroll
  for (int n = 0; n < 8; ++n) {
    A[n] = -expf(Alog[((long long)k * di + d) * 8 + n]);
    h[n] = 0.f;
  }
  float Dv = Ds[k * di + d];
  for (int l = 0; l < L; ++l) {
    float x = xp[l];
    float dt = dp[l];
    float dx = dt * x;
    float y = 0.f;
#pragma unroll
    for (int n = 0; n < 8; ++n) {
      h[n] = h[n] * expf(dt * A[n]) + dx * Bs[(long long)n * L + l];
      y += h[n] * Cs[(long long)n * L + l];
    }
    yp[l] = y + Dv * x;
  }
}

// Combine 4 scan directions -> tokens-major yT[tok, di]
__global__ void dk_combine(const float* __restrict__ ys, float* __restrict__ yT,
                           int B, int di, int H, int W) {
  int n = B * di * H * W;
  int i = blockIdx.x * blockDim.x + threadIdx.x;
  if (i >= n) return;
  int L = H * W;
  int w = i % W, e = i / W;
  int h = e % H; e /= H;
  int c = e % di; e /= di;
  int b = e;
  int l0 = h * W + w;
  int l1 = w * H + h;
  long long base = ((long long)b * 4) * di * L + (long long)c * L;
  long long stride = (long long)di * L;
  float v = ys[base + l0] + ys[base + stride + l1] + ys[base + 2 * stride + (L - 1 - l0)] +
            ys[base + 3 * stride + (L - 1 - l1)];
  yT[((long long)b * L + l0) * di + c] = v;
}

// Build stacked/concatenated vss inputs (modes 1..5); src_i each [B,64,32,32]
__global__ void dk_build_ct(const float* __restrict__ s0, const float* __restrict__ s1,
                            const float* __restrict__ s2, float* __restrict__ ct,
                            int B, int mode) {
  int n = 3 * B * 64 * 32 * 32;
  int i = blockIdx.x * blockDim.x + threadIdx.x;
  if (i >= n) return;
  int w = i % 32, e = i / 32;
  int h = e % 32; e /= 32;
  int c = e % 64; e /= 64;
  int b = e % B; e /= B;
  int src = e;  // 0..2
  const float* sp = (src == 0) ? s0 : (src == 1) ? s1 : s2;
  float v = sp[(((long long)b * 64 + c) * 32 + h) * 32 + w];
  long long o;
  if (mode == 1)      o = (((long long)b * 192 + src * 64 + c) * 32 + h) * 32 + w;
  else if (mode == 2) o = (((long long)b * 64 + c) * 32 + h) * 96 + 3 * w + src;
  else if (mode == 3) o = (((long long)b * 64 + c) * 32 + w) * 96 + 3 * h + src;
  else if (mode == 4) o = (((long long)b * 64 + c) * 32 + h) * 96 + w + 32 * src;
  else                o = (((long long)b * 64 + c) * 32 + w) * 96 + h + 32 * src;
  ct[o] = v;
}

// Extract 3 parts from vss output into the three 320-channel concat buffers
__global__ void dk_extract(const float* __restrict__ p, float* __restrict__ cats,
                           float* __restrict__ cat1, float* __restrict__ cat2,
                           int B, int mode, int choff) {
  int n = 3 * B * 64 * 32 * 32;
  int i = blockIdx.x * blockDim.x + threadIdx.x;
  if (i >= n) return;
  int w = i % 32, e = i / 32;
  int h = e % 32; e /= 32;
  int c = e % 64; e /= 64;
  int b = e % B; e /= B;
  int part = e;  // 0: x_sub slot, 1: x1 slot, 2: x2 slot
  long long o;
  if (mode == 1)      o = (((long long)b * 192 + part * 64 + c) * 32 + h) * 32 + w;
  else if (mode == 2) o = (((long long)b * 64 + c) * 32 + h) * 96 + 3 * w + part;
  else if (mode == 3) o = (((long long)b * 64 + c) * 32 + w) * 96 + 3 * h + part;
  else if (mode == 4) o = (((long long)b * 64 + c) * 32 + h) * 96 + w + 32 * part;
  else                o = (((long long)b * 64 + c) * 32 + w) * 96 + h + 32 * part;
  float v = p[o];
  float* dst = (part == 0) ? cats : (part == 1) ? cat1 : cat2;
  dst[(((long long)b * 320 + choff + c) * 32 + h) * 32 + w] = v;
}

__global__ void dk_absdiff(const float* a, const float* b, float* o, int n) {
  int i = blockIdx.x * blockDim.x + threadIdx.x;
  if (i < n) o[i] = fabsf(a[i] - b[i]);
}
__global__ void dk_add(const float* a, const float* b, float* o, int n) {
  int i = blockIdx.x * blockDim.x + threadIdx.x;
  if (i < n) o[i] = a[i] + b[i];
}

// Full 3x3 conv (SAME) + BN + ReLU
__global__ void dk_conv3_bnrelu(const float* __restrict__ x, const float* __restrict__ wgt,
                                const float* __restrict__ scale, const float* __restrict__ shift,
                                float* __restrict__ y, int B, int Co, int Ci, int H, int W) {
  int n = B * Co * H * W;
  int i = blockIdx.x * blockDim.x + threadIdx.x;
  if (i >= n) return;
  int w = i % W, e = i / W;
  int h = e % H; e /= H;
  int o = e % Co; e /= Co;
  int b = e;
  float acc = 0.f;
  for (int c = 0; c < Ci; ++c) {
    const float* xp = x + ((long long)(b * Ci + c) * H) * W;
    const float* wp = wgt + ((long long)(o * Ci + c)) * 9;
#pragma unroll
    for (int ky = 0; ky < 3; ++ky) {
      int hh = h + ky - 1;
      if (hh < 0 || hh >= H) continue;
#pragma unroll
      for (int kx = 0; kx < 3; ++kx) {
        int ww = w + kx - 1;
        if (ww < 0 || ww >= W) continue;
        acc += xp[hh * W + ww] * wp[ky * 3 + kx];
      }
    }
  }
  y[i] = fmaxf(acc * scale[o] + shift[o], 0.f);
}

// ---------------------------------------------------------------------------
// Host side
// ---------------------------------------------------------------------------
struct Arena {
  char* base;
  size_t off;
  float* alloc(size_t nfloats) {
    size_t bytes = (nfloats * sizeof(float) + 255) & ~(size_t)255;
    float* p = (float*)(base + off);
    off += bytes;
    return p;
  }
};

struct VssP {
  const float *ln1_g, *ln1_b, *in_proj, *conv_w, *conv_b, *x_proj_w, *dt_w, *dt_b,
      *A_log, *Ds, *out_ln_g, *out_ln_b, *out_proj, *ln2_g, *ln2_b, *fc1, *fc1_b,
      *fc2, *fc2_b;
};

static inline int ceildiv(int a, int b) { return (a + b - 1) / b; }

static void launch_gemm(hipStream_t s, const float* A, int lda, long long sAz, int azm,
                        const float* B, int ldb, long long sBz, float* C, int ldc,
                        long long sCz, int M, int N, int K, int Z, int ep,
                        const float* p0, const float* p1, const float* add, int ldadd,
                        long long sAddz) {
  dim3 g(ceildiv(N, GBN), ceildiv(M, GBM), Z);
  dk_gemm<<<g, dim3(128), 0, s>>>(A, lda, sAz, azm, B, ldb, sBz, C, ldc, sCz, M, N, K, ep,
                                  p0, p1, add, ldadd, sAddz);
}

#define EW(n) dim3(ceildiv((int)(n), 256)), dim3(256), 0, stream

static void run_vss(hipStream_t stream, Arena ar, const float* xin_nchw, float* xout_nchw,
                    const VssP& p, int B, int d, int H, int W) {
  const int di = 2 * d;
  const int R = (d / 16 > 1) ? d / 16 : 1;
  const int Nst = 8;
  const int Cdim = R + 2 * Nst;
  const int L = H * W;
  const int T = B * L;

  float* xt   = ar.alloc((size_t)T * d);
  float* xln  = ar.alloc((size_t)T * d);
  float* xz   = ar.alloc((size_t)T * 2 * di);
  float* xcn  = ar.alloc((size_t)B * di * L);
  float* xcv  = ar.alloc((size_t)B * di * L);
  float* xs   = ar.alloc((size_t)B * 4 * di * L);
  float* xdbl = ar.alloc((size_t)B * 4 * Cdim * L);
  float* dts  = ar.alloc((size_t)B * 4 * di * L);
  float* ys   = ar.alloc((size_t)B * 4 * di * L);
  float* yT   = ar.alloc((size_t)T * di);
  float* yg   = ar.alloc((size_t)T * di);
  float* xres = ar.alloc((size_t)T * d);
  float* x2ln = ar.alloc((size_t)T * d);
  float* htok = ar.alloc((size_t)T * 4 * d);
  float* out2 = ar.alloc((size_t)T * d);

  // 1. NCHW -> tokens, ln1
  dk_nchw_to_tok<<<EW((size_t)B * d * L)>>>(xin_nchw, xt, B, d, H, W);
  dk_layernorm<<<dim3(ceildiv(T, 8)), dim3(256), 0, stream>>>(xt, xln, T, d, p.ln1_g,
                                                              p.ln1_b, nullptr, 0, 0);
  // 2. in_proj GEMM: [T,d] x [d,2di]
  launch_gemm(stream, xln, d, 0, 0, p.in_proj, 2 * di, 0, xz, 2 * di, 0, T, 2 * di, d, 1,
              EP_NONE, nullptr, nullptr, nullptr, 0, 0);
  // 3. split -> NCHW, depthwise conv + silu
  dk_split_nchw<<<EW((size_t)B * di * L)>>>(xz, xcn, B, di, H, W);
  dk_dwconv_silu<<<EW((size_t)B * di * L)>>>(xcn, p.conv_w, p.conv_b, xcv, B, di, H, W);
  // 4. four scan directions
  dk_build_xs<<<EW((size_t)B * di * L)>>>(xcv, xs, B, di, H, W);
  // 5. x_proj: per (b,k): [Cdim,di] x [di,L]
  launch_gemm(stream, p.x_proj_w, di, (long long)Cdim * di, 4, xs, L, (long long)di * L,
              xdbl, L, (long long)Cdim * L, Cdim, L, di, B * 4, EP_NONE, nullptr, nullptr,
              nullptr, 0, 0);
  // 6. dt proj: per (b,k): [di,R] x [R,L]
  launch_gemm(stream, p.dt_w, R, (long long)di * R, 4, xdbl, L, (long long)Cdim * L, dts,
              L, (long long)di * L, di, L, R, B * 4, EP_NONE, nullptr, nullptr, nullptr,
              0, 0);
  // 7. softplus(dts + dt_b)
  dk_softplus_dt<<<EW((size_t)B * 4 * di * L)>>>(dts, p.dt_b, B, di, L);
  // 8. selective scan
  dk_scan<<<EW(B * 4 * di)>>>(xs, dts, xdbl, p.A_log, p.Ds, ys, B, di, L, R, Cdim);
  // 9. combine directions -> tokens
  dk_combine<<<EW((size_t)B * di * L)>>>(ys, yT, B, di, H, W);
  // 10. out layernorm with fused silu(z) gate
  dk_layernorm<<<dim3(ceildiv(T, 8)), dim3(256), 0, stream>>>(yT, yg, T, di, p.out_ln_g,
                                                              p.out_ln_b, xz, 2 * di, di);
  // 11. out_proj + residual (x = x + ss2d(ln1(x)))
  launch_gemm(stream, yg, di, 0, 0, p.out_proj, d, 0, xres, d, 0, T, d, di, 1, EP_ADD,
              nullptr, nullptr, xt, d, 0);
  // 12. ln2, fc1+gelu, fc2+residual
  dk_layernorm<<<dim3(ceildiv(T, 8)), dim3(256), 0, stream>>>(xres, x2ln, T, d, p.ln2_g,
                                                              p.ln2_b, nullptr, 0, 0);
  launch_gemm(stream, x2ln, d, 0, 0, p.fc1, 4 * d, 0, htok, 4 * d, 0, T, 4 * d, d, 1,
              EP_BIAS_GELU, p.fc1_b, nullptr, nullptr, 0, 0);
  launch_gemm(stream, htok, 4 * d, 0, 0, p.fc2, d, 0, out2, d, 0, T, d, 4 * d, 1,
              EP_BIAS_ADD, p.fc2_b, nullptr, xres, d, 0);
  // 13. tokens -> NCHW
  dk_tok_to_nchw<<<EW((size_t)B * d * L)>>>(out2, xout_nchw, B, d, H, W);
}

extern "C" void kernel_launch(void* const* d_in, const int* in_sizes, int n_in,
                              void* d_out, int out_size, void* d_ws, size_t ws_size,
                              hipStream_t stream) {
  (void)in_sizes; (void)n_in; (void)out_size; (void)ws_size;
  const int B = 2, Cin = 128, Cc = 64, H = 32, W = 32, HW = H * W;
  auto P = [&](int i) { return (const float*)d_in[i]; };

  const float* x1 = P(0);
  const float* x2 = P(1);
  const float* x_sub = P(2);
  int idx = 3;
  const float* in1_w = P(idx++);
  const float *in1_g = P(idx++), *in1_b = P(idx++), *in1_m = P(idx++), *in1_v = P(idx++);
  const float* in2_w = P(idx++);
  const float *in2_g = P(idx++), *in2_b = P(idx++), *in2_m = P(idx++), *in2_v = P(idx++);
  VssP vp[5];
  for (int v = 0; v < 5; ++v) {
    vp[v].ln1_g = P(idx++);    vp[v].ln1_b = P(idx++);    vp[v].in_proj = P(idx++);
    vp[v].conv_w = P(idx++);   vp[v].conv_b = P(idx++);   vp[v].x_proj_w = P(idx++);
    vp[v].dt_w = P(idx++);     vp[v].dt_b = P(idx++);     vp[v].A_log = P(idx++);
    vp[v].Ds = P(idx++);       vp[v].out_ln_g = P(idx++); vp[v].out_ln_b = P(idx++);
    vp[v].out_proj = P(idx++); vp[v].ln2_g = P(idx++);    vp[v].ln2_b = P(idx++);
    vp[v].fc1 = P(idx++);      vp[v].fc1_b = P(idx++);    vp[v].fc2 = P(idx++);
    vp[v].fc2_b = P(idx++);
  }
  const float* fuse1_w = P(idx++);
  const float *f1g = P(idx++), *f1b = P(idx++), *f1m = P(idx++), *f1v = P(idx++);
  const float* fuse2_w = P(idx++);
  const float *f2g = P(idx++), *f2b = P(idx++), *f2m = P(idx++), *f2v = P(idx++);
  const float* fuses_w = P(idx++);
  const float *fsg = P(idx++), *fsb = P(idx++), *fsm = P(idx++), *fsv = P(idx++);
  const float* sub_w = P(idx++);
  const float *sbg = P(idx++), *sbb = P(idx++), *sbm = P(idx++), *sbv = P(idx++);
  const float* out_w = P(idx++);
  const float *og = P(idx++), *ob = P(idx++), *om = P(idx++), *ov = P(idx++);

  Arena ar{(char*)d_ws, 0};
  // Persistent buffers
  float* bnbuf = ar.alloc(7 * 128);  // 7 x (scale[64], shift[64])
  float* x1c = ar.alloc((size_t)B * Cc * HW);
  float* x2c = ar.alloc((size_t)B * Cc * HW);
  float* cat1 = ar.alloc((size_t)B * 320 * HW);
  float* cat2 = ar.alloc((size_t)B * 320 * HW);
  float* cats = ar.alloc((size_t)B * 320 * HW);
  float* p_1 = ar.alloc((size_t)B * Cc * HW);
  float* p_2 = ar.alloc((size_t)B * Cc * HW);
  float* p_s = ar.alloc((size_t)B * Cc * HW);
  float* tmpA = ar.alloc((size_t)B * Cc * HW);
  float* tmpB = ar.alloc((size_t)B * Cc * HW);
  float* ctbuf = ar.alloc((size_t)B * 192 * HW);  // also fits [B,64,32,96]
  float* pbuf = ar.alloc((size_t)B * 192 * HW);
  // Everything after this point is scratch re-used per VSS block.
  Arena scratch = ar;

  float* sc[7];
  float* sh[7];
  const float* bg[7] = {in1_g, in2_g, f1g, f2g, fsg, sbg, og};
  const float* bb[7] = {in1_b, in2_b, f1b, f2b, fsb, sbb, ob};
  const float* bm[7] = {in1_m, in2_m, f1m, f2m, fsm, sbm, om};
  const float* bv[7] = {in1_v, in2_v, f1v, f2v, fsv, sbv, ov};
  for (int i = 0; i < 7; ++i) {
    sc[i] = bnbuf + i * 128;
    sh[i] = bnbuf + i * 128 + 64;
    dk_bnprep<<<dim3(1), dim3(64), 0, stream>>>(bg[i], bb[i], bm[i], bv[i], sc[i], sh[i], 64);
  }

  // conv1x1 + bn + relu on x1, x2 (GEMM per batch: W[64x128] * X[128xHW])
  launch_gemm(stream, in1_w, Cin, 0, 0, x1, HW, (long long)Cin * HW, x1c, HW,
              (long long)Cc * HW, Cc, HW, Cin, B, EP_BNRELU, sc[0], sh[0], nullptr, 0, 0);
  launch_gemm(stream, in2_w, Cin, 0, 0, x2, HW, (long long)Cin * HW, x2c, HW,
              (long long)Cc * HW, Cc, HW, Cin, B, EP_BNRELU, sc[1], sh[1], nullptr, 0, 0);

  // VSS blocks 1..5
  for (int m = 1; m <= 5; ++m) {
    dk_build_ct<<<EW(3 * B * Cc * HW)>>>(x_sub, x1c, x2c, ctbuf, B, m);
    if (m == 1)
      run_vss(stream, scratch, ctbuf, pbuf, vp[0], B, 192, 32, 32);
    else
      run_vss(stream, scratch, ctbuf, pbuf, vp[m - 1], B, 64, 32, 96);
    dk_extract<<<EW(3 * B * Cc * HW)>>>(pbuf, cats, cat1, cat2, B, m, (m - 1) * 64);
  }

  // Fuse conv1x1s (K=320) with bn+relu
  launch_gemm(stream, fuse1_w, 320, 0, 0, cat1, HW, (long long)320 * HW, p_1, HW,
              (long long)Cc * HW, Cc, HW, 320, B, EP_BNRELU, sc[2], sh[2], nullptr, 0, 0);
  launch_gemm(stream, fuse2_w, 320, 0, 0, cat2, HW, (long long)320 * HW, p_2, HW,
              (long long)Cc * HW, Cc, HW, 320, B, EP_BNRELU, sc[3], sh[3], nullptr, 0, 0);
  launch_gemm(stream, fuses_w, 320, 0, 0, cats, HW, (long long)320 * HW, p_s, HW,
              (long long)Cc * HW, Cc, HW, 320, B, EP_BNRELU, sc[4], sh[4], nullptr, 0, 0);

  int nelem = B * Cc * HW;
  dk_absdiff<<<EW(nelem)>>>(p_1, p_2, tmpA, nelem);
  dk_conv3_bnrelu<<<EW(nelem)>>>(tmpA, sub_w, sc[5], sh[5], tmpB, B, Cc, Cc, H, W);
  dk_add<<<EW(nelem)>>>(tmpB, p_s, tmpA, nelem);
  dk_conv3_bnrelu<<<EW(nelem)>>>(tmpA, out_w, sc[6], sh[6], (float*)d_out, B, Cc, Cc, H, W);
}
